// NonLocalBlock_90563680403808
// MI455X (gfx1250) — matrix operand order repaired and checked by reading, compiled, not run on hardware
//
#include <hip/hip_runtime.h>
#include <hip/hip_bf16.h>
#include <cstdint>

// ---------------------------------------------------------------------------
// NonLocal attention block for MI455X (gfx1250), wave32 + WMMA f16->f32.
// Shapes: B=16, C=512, H=W=48 -> N=2304, 32 groups GN.
// GEMM kernels use 2x2 register blocking + double-buffered K-loop so WMMAs
// overlap the next K-step's global loads (partial s_wait_loadcnt).
// ---------------------------------------------------------------------------

typedef _Float16 h8v  __attribute__((ext_vector_type(8)));
typedef _Float16 h16v __attribute__((ext_vector_type(16)));
typedef float    f8v  __attribute__((ext_vector_type(8)));

#define BB 16
#define CC 512
#define NN 2304
#define NG 32

// ---- WMMA operand loaders (layouts per CDNA5 ISA 7.12.2, wave32) ----------
// A-matrix 16x32 f16: lane L -> row = row0 + L%16 ; lane-half g = L/16:
//   halves 0..7  = K  k0 + 8g + {0..7} ; halves 8..15 = K k0 + 16 + 8g + {0..7}
static __device__ __forceinline__ h16v load_amat(const _Float16* __restrict__ base,
                                                 int row0, int k0, int pitch, int lane) {
  const _Float16* p = base + (size_t)(row0 + (lane & 15)) * (size_t)pitch
                           + (size_t)(k0 + ((lane >> 4) << 3));
  h8v lo = *reinterpret_cast<const h8v*>(p);
  h8v hi = *reinterpret_cast<const h8v*>(p + 16);
  return __builtin_shufflevector(lo, hi, 0,1,2,3,4,5,6,7,8,9,10,11,12,13,14,15);
}

// B-matrix 32x16 f16: lane L -> col = col0 + L%16 ; lane-half g = L/16:
//   halves 0..7  = K  k0 + 8g + {0..7} ; halves 8..15 = K k0 + 16 + 8g + {0..7}
static __device__ __forceinline__ h16v load_bmat(const _Float16* __restrict__ base,
                                                 int col0, int k0, int pitch, int lane) {
  const _Float16* p = base + (size_t)(col0 + (lane & 15)) * (size_t)pitch
                           + (size_t)(k0 + ((lane >> 4) << 3));
  h8v lo = *reinterpret_cast<const h8v*>(p);
  h8v hi = *reinterpret_cast<const h8v*>(p + 16);
  return __builtin_shufflevector(lo, hi, 0,1,2,3,4,5,6,7,8,9,10,11,12,13,14,15);
}

static __device__ __forceinline__ f8v wmma16(h16v a, h16v b, f8v c) {
  return __builtin_amdgcn_wmma_f32_16x16x32_f16(false, a, false, b, (short)0, c, false, false);
}

// ---------------------------------------------------------------------------
// Weight conversion f32 -> f16
// ---------------------------------------------------------------------------
__global__ void f32_to_f16_kernel(const float* __restrict__ src,
                                  _Float16* __restrict__ dst, int n) {
  int i = blockIdx.x * 256 + threadIdx.x;
  if (i < n) dst[i] = (_Float16)src[i];
}

// ---------------------------------------------------------------------------
// GroupNorm stats: one workgroup per (b, g). 16 channels x N elements.
// ---------------------------------------------------------------------------
__global__ void gn_stats_kernel(const float* __restrict__ x,
                                float2* __restrict__ stats) {
  const int b = blockIdx.x >> 5;
  const int g = blockIdx.x & 31;
  const float* base = x + ((size_t)b * CC + (size_t)g * 16) * NN;
  const int total = 16 * NN;
  float s = 0.f, s2 = 0.f;
  for (int idx = threadIdx.x; idx < total; idx += 256) {
    float v = base[idx];
    s += v; s2 += v * v;
  }
  __shared__ float rs[256];
  __shared__ float rq[256];
  int t = threadIdx.x;
  rs[t] = s; rq[t] = s2;
  __syncthreads();
  for (int st = 128; st > 0; st >>= 1) {
    if (t < st) { rs[t] += rs[t + st]; rq[t] += rq[t + st]; }
    __syncthreads();
  }
  if (t == 0) {
    float invn = 1.0f / (float)total;
    float mean = rs[0] * invn;
    float var  = rq[0] * invn - mean * mean;
    stats[blockIdx.x] = make_float2(mean, rsqrtf(var + 1e-5f));
  }
}

// ---------------------------------------------------------------------------
// GroupNorm apply + transpose: x[B][C][N] f32 -> y_t[B][N][C] f16
// ---------------------------------------------------------------------------
__global__ void gn_apply_kernel(const float* __restrict__ x,
                                const float* __restrict__ gamma,
                                const float* __restrict__ beta,
                                const float2* __restrict__ stats,
                                _Float16* __restrict__ yt) {
  const int b  = blockIdx.z;
  const int n0 = blockIdx.x * 64;
  const int c0 = blockIdx.y * 64;
  __shared__ _Float16 tile[64][72];   // +8 pad vs bank conflicts

  const int t  = threadIdx.x;
  {
    const int cl = t >> 2;              // 0..63 channel-in-tile
    const int nb = (t & 3) * 16;        // 16 n per thread
    const int c  = c0 + cl;
    const int g  = c >> 4;
    float2 st = stats[b * NG + g];
    float ga = gamma[c] * st.y;
    float be = beta[c] - st.x * st.y * gamma[c];
    const float* xp = x + ((size_t)b * CC + c) * NN + n0 + nb;
#pragma unroll
    for (int u = 0; u < 16; ++u)
      tile[cl][nb + u] = (_Float16)(xp[u] * ga + be);
  }
  __syncthreads();
  {
    const int nl = t >> 2;              // 0..63 position-in-tile
    const int cb = (t & 3) * 16;        // 16 c per thread
    h8v o0, o1;
#pragma unroll
    for (int u = 0; u < 8; ++u) { o0[u] = tile[cb + u][nl]; o1[u] = tile[cb + 8 + u][nl]; }
    _Float16* yp = yt + ((size_t)b * NN + n0 + nl) * CC + c0 + cb;
    *reinterpret_cast<h8v*>(yp)     = o0;
    *reinterpret_cast<h8v*>(yp + 8) = o1;
  }
}

// ---------------------------------------------------------------------------
// Projection GEMM: D[o][pos] = sum_c W[o][c] * Y[b][pos][c]  (+bias)
// Block tile: o 128 x pos 64. Wave: 2 o-subs x 2 pos-subs, double-buffered K.
// MODE 0: f16 -> out_t[b][pos][o] (packed 8xf16)    -> q_t, k_t
// MODE 1: f16 -> out[b][o][pos]                     -> v_cn
// MODE 2: f32 -> out[b][o][pos] + bias + residual x -> d_out
// ---------------------------------------------------------------------------
template<int MODE>
__global__ __launch_bounds__(256)
void proj_gemm_kernel(const _Float16* __restrict__ W,   // [512][512] f16
                      const _Float16* __restrict__ Y,   // [B][N][512] f16
                      const float* __restrict__ bias,   // [512]
                      const float* __restrict__ resid,  // x [B][512][N] (MODE 2)
                      void* __restrict__ outp) {
  const int b    = blockIdx.z;
  const int pos0 = blockIdx.x * 64;
  const int ob0  = blockIdx.y * 128;
  const int lane = threadIdx.x & 31;
  const int wave = threadIdx.x >> 5;
  const int l16  = lane & 15;
  const int lhi  = lane >> 4;

  const int mSub = ob0 + (wave & 3) * 32;          // A rows: mSub, mSub+16
  const int nSub = pos0 + (wave >> 2) * 32;        // B cols: nSub, nSub+16

  const _Float16* Yb = Y + (size_t)b * NN * CC;

  f8v acc[4] = {};                                 // [mi*2+ni]
  h16v a0 = load_amat(W,  mSub,      0, CC, lane);
  h16v a1 = load_amat(W,  mSub + 16, 0, CC, lane);
  h16v b0 = load_bmat(Yb, nSub,      0, CC, lane);
  h16v b1 = load_bmat(Yb, nSub + 16, 0, CC, lane);
  for (int k0 = 32; k0 < CC; k0 += 32) {
    h16v na0 = load_amat(W,  mSub,      k0, CC, lane);
    h16v na1 = load_amat(W,  mSub + 16, k0, CC, lane);
    h16v nb0 = load_bmat(Yb, nSub,      k0, CC, lane);
    h16v nb1 = load_bmat(Yb, nSub + 16, k0, CC, lane);
    acc[0] = wmma16(a0, b0, acc[0]);
    acc[1] = wmma16(a0, b1, acc[1]);
    acc[2] = wmma16(a1, b0, acc[2]);
    acc[3] = wmma16(a1, b1, acc[3]);
    a0 = na0; a1 = na1; b0 = nb0; b1 = nb1;
  }
  acc[0] = wmma16(a0, b0, acc[0]);
  acc[1] = wmma16(a0, b1, acc[1]);
  acc[2] = wmma16(a1, b0, acc[2]);
  acc[3] = wmma16(a1, b1, acc[3]);

#pragma unroll
  for (int mi = 0; mi < 2; ++mi) {
    const int ob = mSub + mi * 16 + 8 * lhi;       // 8 contiguous o per lane
#pragma unroll
    for (int ni = 0; ni < 2; ++ni) {
      const int pos = nSub + ni * 16 + l16;
      const f8v a = acc[mi * 2 + ni];
      if (MODE == 0) {
        float4 c0 = *reinterpret_cast<const float4*>(bias + ob);
        float4 c1 = *reinterpret_cast<const float4*>(bias + ob + 4);
        h8v hv;
        hv[0] = (_Float16)(a[0] + c0.x); hv[1] = (_Float16)(a[1] + c0.y);
        hv[2] = (_Float16)(a[2] + c0.z); hv[3] = (_Float16)(a[3] + c0.w);
        hv[4] = (_Float16)(a[4] + c1.x); hv[5] = (_Float16)(a[5] + c1.y);
        hv[6] = (_Float16)(a[6] + c1.z); hv[7] = (_Float16)(a[7] + c1.w);
        _Float16* O = (_Float16*)outp;
        *reinterpret_cast<h8v*>(O + ((size_t)b * NN + pos) * CC + ob) = hv;
      } else if (MODE == 1) {
        _Float16* O = (_Float16*)outp;
#pragma unroll
        for (int r = 0; r < 8; ++r)
          O[((size_t)b * CC + ob + r) * NN + pos] = (_Float16)(a[r] + bias[ob + r]);
      } else {
        float* O = (float*)outp;
#pragma unroll
        for (int r = 0; r < 8; ++r) {
          size_t ad = ((size_t)b * CC + ob + r) * (size_t)NN + pos;
          O[ad] = a[r] + bias[ob + r] + resid[ad];
        }
      }
    }
  }
}

// ---------------------------------------------------------------------------
// Attention scores: S[b][i][j] = (1/sqrt(C)) sum_c q_t[b][i][c] * k_t[b][j][c]
// Block tile: i 64 x j 128. Wave: 2 i-subs x 2 j-subs, double-buffered K.
// ---------------------------------------------------------------------------
__global__ __launch_bounds__(256)
void attn_scores_kernel(const _Float16* __restrict__ Q,   // [B][N][512]
                        const _Float16* __restrict__ K,   // [B][N][512]
                        float* __restrict__ S) {          // [B][N][N]
  const int b  = blockIdx.z;
  const int i0 = blockIdx.x * 64;
  const int j0 = blockIdx.y * 128;
  const int lane = threadIdx.x & 31;
  const int wave = threadIdx.x >> 5;
  const int l16 = lane & 15, lhi = lane >> 4;

  const int iSub = i0 + (wave & 1) * 32;           // A rows: iSub, iSub+16
  const int jSub = j0 + (wave >> 1) * 32;          // B cols: jSub, jSub+16

  const _Float16* Qb = Q + (size_t)b * NN * CC;
  const _Float16* Kb = K + (size_t)b * NN * CC;

  f8v acc[4] = {};
  h16v a0 = load_amat(Qb, iSub,      0, CC, lane);
  h16v a1 = load_amat(Qb, iSub + 16, 0, CC, lane);
  h16v b0 = load_bmat(Kb, jSub,      0, CC, lane);
  h16v b1 = load_bmat(Kb, jSub + 16, 0, CC, lane);
  for (int k0 = 32; k0 < CC; k0 += 32) {
    h16v na0 = load_amat(Qb, iSub,      k0, CC, lane);
    h16v na1 = load_amat(Qb, iSub + 16, k0, CC, lane);
    h16v nb0 = load_bmat(Kb, jSub,      k0, CC, lane);
    h16v nb1 = load_bmat(Kb, jSub + 16, k0, CC, lane);
    acc[0] = wmma16(a0, b0, acc[0]);
    acc[1] = wmma16(a0, b1, acc[1]);
    acc[2] = wmma16(a1, b0, acc[2]);
    acc[3] = wmma16(a1, b1, acc[3]);
    a0 = na0; a1 = na1; b0 = nb0; b1 = nb1;
  }
  acc[0] = wmma16(a0, b0, acc[0]);
  acc[1] = wmma16(a0, b1, acc[1]);
  acc[2] = wmma16(a1, b0, acc[2]);
  acc[3] = wmma16(a1, b1, acc[3]);

  const float scale = 0.044194173824159216f;       // 1/sqrt(512)
  float* Sb = S + (size_t)b * NN * NN;
#pragma unroll
  for (int mi = 0; mi < 2; ++mi) {
#pragma unroll
    for (int ni = 0; ni < 2; ++ni) {
      const int j = jSub + ni * 16 + l16;
      const f8v a = acc[mi * 2 + ni];
#pragma unroll
      for (int r = 0; r < 8; ++r) {
        const int i = iSub + mi * 16 + 8 * lhi + r;
        Sb[(size_t)i * NN + j] = a[r] * scale;
      }
    }
  }
}

// ---------------------------------------------------------------------------
// Softmax over each S row (length N); writes P = softmax(S) as f16 in-place
// into the front half of the f32 row. Reads precede writes (barrier-ordered).
// ---------------------------------------------------------------------------
__global__ void softmax_kernel(float* __restrict__ S) {
  float* p = S + (size_t)blockIdx.x * NN;          // blockIdx.x = b*N + i
  const int t = threadIdx.x;
  float v[9];
  float m = -3.4e38f;
#pragma unroll
  for (int u = 0; u < 9; ++u) { v[u] = p[t * 9 + u]; m = fmaxf(m, v[u]); }
  __shared__ float red[256];
  red[t] = m;
  __syncthreads();
  for (int st = 128; st > 0; st >>= 1) {
    if (t < st) red[t] = fmaxf(red[t], red[t + st]);
    __syncthreads();
  }
  const float mx = red[0];
  __syncthreads();
  float s = 0.f;
#pragma unroll
  for (int u = 0; u < 9; ++u) { v[u] = __expf(v[u] - mx); s += v[u]; }
  red[t] = s;
  __syncthreads();
  for (int st = 128; st > 0; st >>= 1) {
    if (t < st) red[t] += red[t + st];
    __syncthreads();
  }
  const float inv = 1.0f / red[0];
  _Float16* ph = (_Float16*)p;
#pragma unroll
  for (int u = 0; u < 9; ++u) ph[t * 9 + u] = (_Float16)(v[u] * inv);
}

// ---------------------------------------------------------------------------
// PV GEMM: o_t[b][i][c] = sum_j v_cn[b][c][j] * P[b][i][j]
// A = v_cn (m=c, K=j contiguous), B = P (n=i, K=j contiguous, pitch 2N halfs)
// Block tile: c 128 x i 64. Wave: 2 c-subs x 2 i-subs, double-buffered K.
// ---------------------------------------------------------------------------
__global__ __launch_bounds__(256)
void attn_pv_kernel(const _Float16* __restrict__ V,  // [B][512][N] f16
                    const float* __restrict__ S,     // rows hold f16 P in front half
                    _Float16* __restrict__ Ot) {     // [B][N][512] f16
  const int b  = blockIdx.z;
  const int i0 = blockIdx.x * 64;
  const int c0 = blockIdx.y * 128;
  const int lane = threadIdx.x & 31;
  const int wave = threadIdx.x >> 5;
  const int l16 = lane & 15, lhi = lane >> 4;

  const int cSub = c0 + (wave & 3) * 32;           // A rows: cSub, cSub+16
  const int iSub = i0 + (wave >> 2) * 32;          // B cols: iSub, iSub+16

  const _Float16* Pb = (const _Float16*)(S + (size_t)b * NN * NN); // pitch 2*NN halfs
  const _Float16* Vb = V + (size_t)b * CC * NN;

  f8v acc[4] = {};
  h16v a0 = load_amat(Vb, cSub,      0, NN, lane);
  h16v a1 = load_amat(Vb, cSub + 16, 0, NN, lane);
  h16v b0 = load_bmat(Pb, iSub,      0, 2 * NN, lane);
  h16v b1 = load_bmat(Pb, iSub + 16, 0, 2 * NN, lane);
  for (int k0 = 32; k0 < NN; k0 += 32) {
    h16v na0 = load_amat(Vb, cSub,      k0, NN, lane);
    h16v na1 = load_amat(Vb, cSub + 16, k0, NN, lane);
    h16v nb0 = load_bmat(Pb, iSub,      k0, 2 * NN, lane);
    h16v nb1 = load_bmat(Pb, iSub + 16, k0, 2 * NN, lane);
    acc[0] = wmma16(a0, b0, acc[0]);
    acc[1] = wmma16(a0, b1, acc[1]);
    acc[2] = wmma16(a1, b0, acc[2]);
    acc[3] = wmma16(a1, b1, acc[3]);
    a0 = na0; a1 = na1; b0 = nb0; b1 = nb1;
  }
  acc[0] = wmma16(a0, b0, acc[0]);
  acc[1] = wmma16(a0, b1, acc[1]);
  acc[2] = wmma16(a1, b0, acc[2]);
  acc[3] = wmma16(a1, b1, acc[3]);

#pragma unroll
  for (int mi = 0; mi < 2; ++mi) {
    const int cbse = cSub + mi * 16 + 8 * lhi;
#pragma unroll
    for (int ni = 0; ni < 2; ++ni) {
      const int i = iSub + ni * 16 + l16;
      const f8v a = acc[mi * 2 + ni];
      h8v hv;
#pragma unroll
      for (int r = 0; r < 8; ++r) hv[r] = (_Float16)a[r];
      *reinterpret_cast<h8v*>(Ot + ((size_t)b * NN + i) * CC + cbse) = hv;
    }
  }
}

// ---------------------------------------------------------------------------
// Host-side orchestration
// ---------------------------------------------------------------------------
extern "C" void kernel_launch(void* const* d_in, const int* in_sizes, int n_in,
                              void* d_out, int out_size, void* d_ws, size_t ws_size,
                              hipStream_t stream) {
  const float* x     = (const float*)d_in[0];
  const float* gamma = (const float*)d_in[1];
  const float* beta  = (const float*)d_in[2];
  const float* bq    = (const float*)d_in[4];
  const float* bk    = (const float*)d_in[6];
  const float* bv    = (const float*)d_in[8];
  const float* bo    = (const float*)d_in[10];

  const size_t SZ_S = (size_t)BB * NN * NN * 4;   // 339,738,624
  const size_t SZ_T = (size_t)BB * NN * CC * 2;   //  37,748,736
  const size_t SZ_W = (size_t)CC * CC * 2;        //     524,288

  char* w = (char*)d_ws;
  float*     S    = (float*)w;                          w += SZ_S;
  _Float16*  yt   = (_Float16*)w;                       w += SZ_T;
  _Float16*  qt   = (_Float16*)w;                       w += SZ_T;
  _Float16*  kt   = (_Float16*)w;                       w += SZ_T;
  _Float16*  vcn  = (_Float16*)w;                       w += SZ_T;
  _Float16*  ot   = (_Float16*)w;                       w += SZ_T;
  _Float16*  wh   = (_Float16*)w;                       w += 4 * SZ_W;
  float2*    stats= (float2*)w;

  _Float16* whq = wh;
  _Float16* whk = wh + (size_t)CC * CC;
  _Float16* whv = wh + (size_t)2 * CC * CC;
  _Float16* who = wh + (size_t)3 * CC * CC;

  dim3 blk(256);

  // 1) weights f32 -> f16
  f32_to_f16_kernel<<<dim3(1024), blk, 0, stream>>>((const float*)d_in[3], whq, CC * CC);
  f32_to_f16_kernel<<<dim3(1024), blk, 0, stream>>>((const float*)d_in[5], whk, CC * CC);
  f32_to_f16_kernel<<<dim3(1024), blk, 0, stream>>>((const float*)d_in[7], whv, CC * CC);
  f32_to_f16_kernel<<<dim3(1024), blk, 0, stream>>>((const float*)d_in[9], who, CC * CC);

  // 2) GroupNorm
  gn_stats_kernel<<<dim3(BB * NG), blk, 0, stream>>>(x, stats);
  gn_apply_kernel<<<dim3(NN / 64, CC / 64, BB), blk, 0, stream>>>(x, gamma, beta, stats, yt);

  // 3) Q, K, V projections (WMMA)
  proj_gemm_kernel<0><<<dim3(NN / 64, CC / 128, BB), blk, 0, stream>>>(whq, yt, bq, nullptr, qt);
  proj_gemm_kernel<0><<<dim3(NN / 64, CC / 128, BB), blk, 0, stream>>>(whk, yt, bk, nullptr, kt);
  proj_gemm_kernel<1><<<dim3(NN / 64, CC / 128, BB), blk, 0, stream>>>(whv, yt, bv, nullptr, vcn);

  // 4) attention scores + softmax (in-place f16 P) + PV (WMMA)
  attn_scores_kernel<<<dim3(NN / 64, NN / 128, BB), blk, 0, stream>>>(qt, kt, S);
  softmax_kernel<<<dim3(BB * NN), blk, 0, stream>>>(S);
  attn_pv_kernel<<<dim3(NN / 64, CC / 128, BB), blk, 0, stream>>>(vcn, S, ot);

  // 5) output projection + bias + residual, f32 out
  proj_gemm_kernel<2><<<dim3(NN / 64, CC / 128, BB), blk, 0, stream>>>(who, ot, bo, x, (float*)d_out);
}
